// RelativeMultiheadAttention_9869834847024
// MI455X (gfx1250) — matrix operand order repaired and checked by reading, not compiled, not run on hardware
//
#include <hip/hip_runtime.h>

// ---------------------------------------------------------------------------
// Transformer-XL relative multi-head attention for MI455X (gfx1250, wave32).
// bf16 WMMA everywhere; ALL matrix tiles staged by the Tensor Data Mover
// (tensor_load_to_lds, TENSORcnt double-buffering); flash-style fused softmax;
// rel-shift via diagonal-band WMMA + ds_bpermute extraction.
// ---------------------------------------------------------------------------

typedef __attribute__((ext_vector_type(16))) __bf16 v16bf;
typedef __attribute__((ext_vector_type(8)))  float  v8f;
typedef __attribute__((ext_vector_type(2)))  __bf16 bf16x2;
typedef __attribute__((ext_vector_type(4)))  __bf16 bf16x4;
typedef __attribute__((ext_vector_type(4)))  unsigned int v4u;
typedef __attribute__((ext_vector_type(8)))  int v8i;
typedef __attribute__((ext_vector_type(4)))  int v4i;

#define DIMX   1024
#define HEADSX 16
#define HDIMX  64
#define BX     2
#define TX     2048

__device__ __forceinline__ float shflf(float v, int src) {
  return __int_as_float(__builtin_amdgcn_ds_bpermute(src << 2, __float_as_int(v)));
}
// xor-shuffle via ds_swizzle immediate (group-of-32: and=0x1F, xor=m)
#define SWZ_XOR(v, m) \
  __int_as_float(__builtin_amdgcn_ds_swizzle(__float_as_int(v), ((m) << 10) | 0x1F))

__device__ __forceinline__ v8f wmma_bf16(v16bf a, v16bf b, v8f c) {
  return __builtin_amdgcn_wmma_f32_16x16x32_bf16(false, a, false, b, (short)0, c,
                                                 false, false);
}

// ---------------------------------------------------------------------------
// Tensor Data Mover: 2-D bf16 tile  global -> LDS  (ISA 8.x D# layout).
// Tile lands in LDS row-major as [tile_h][tile_w]. OOB rows/cols read zero.
// ---------------------------------------------------------------------------
__device__ __forceinline__ void tdm_load_2d(void* lds_dst, const void* gsrc,
                                            unsigned row_len, unsigned nrows,
                                            unsigned row_stride,
                                            unsigned tile_w, unsigned tile_h) {
  unsigned long long ga = (unsigned long long)gsrc;
  unsigned lds  = (unsigned)__builtin_amdgcn_readfirstlane(
                      (int)(unsigned)(unsigned long long)lds_dst);
  unsigned galo = (unsigned)__builtin_amdgcn_readfirstlane((int)(unsigned)ga);
  unsigned gahi = (unsigned)__builtin_amdgcn_readfirstlane(
                      (int)(unsigned)(ga >> 32)) & 0x1ffffffu;
  v4u g0;
  g0.x = 1u;                      // count=1 (valid user descriptor)
  g0.y = lds;                     // lds_addr
  g0.z = galo;                    // global_addr[31:0]
  g0.w = gahi | (2u << 30);       // global_addr[56:32] | type=2 ("image")
  v8i g1;
  g1[0] = (int)(1u << 16);        // wg_mask=0, data_size=1 (2 bytes)
  g1[1] = (int)(row_len << 16);                     // tensor_dim0[15:0]
  g1[2] = (int)((row_len >> 16) | (nrows << 16));   // dim0 hi | tensor_dim1 lo
  g1[3] = (int)((nrows >> 16) | (tile_w << 16));    // dim1 hi | tile_dim0
  g1[4] = (int)tile_h;            // tile_dim1 (tile_dim2 = 0)
  g1[5] = (int)row_stride;        // tensor_dim0_stride[31:0]
  g1[6] = 0;
  g1[7] = 0;
  v4i zz = {0, 0, 0, 0};
#if defined(__clang_major__) && (__clang_major__ >= 23)
  v8i z8 = {0, 0, 0, 0, 0, 0, 0, 0};
  __builtin_amdgcn_tensor_load_to_lds(g0, g1, zz, zz, z8, 0);
#else
  __builtin_amdgcn_tensor_load_to_lds(g0, g1, zz, zz, 0);
#endif
}

// ---- fragment builders (CDNA5 ISA 7.12.2 layouts, wave32) -----------------

// A-matrix 16x32 bf16 from row-major [16][stride]  (pairs contiguous in k)
__device__ __forceinline__ v16bf afrag(const __bf16* src, int stride, int kofs, int lane) {
  int m = lane & 15, half = lane >> 4;
  v16bf a;
#pragma unroll
  for (int v2 = 0; v2 < 8; ++v2) {
    int k = ((v2 < 4) ? (2 * v2 + 8 * half) : (16 + 2 * (v2 - 4) + 8 * half)) + kofs;
    bf16x2 p = *(const bf16x2*)&src[(size_t)m * stride + k];
    a[2 * v2] = p.x; a[2 * v2 + 1] = p.y;
  }
  return a;
}

// B-matrix 32x16 bf16 where B[k][n] = base[(col0+n)*stride + k0 + k]
__device__ __forceinline__ v16bf bfragS(const __bf16* base, int stride, int col0,
                                        int k0, int lane) {
  int n = lane & 15, half = lane >> 4;
  v16bf b;
#pragma unroll
  for (int v2 = 0; v2 < 8; ++v2) {
    int k = ((v2 < 4) ? (2 * v2 + 8 * half) : (16 + 2 * (v2 - 4) + 8 * half)) + k0;
    bf16x2 p = *(const bf16x2*)&base[(size_t)(col0 + n) * stride + k];
    b[2 * v2] = p.x; b[2 * v2 + 1] = p.y;
  }
  return b;
}

// ---- generic 16x16 GEMM tile: C = A(bf16 [M][K]) x B^T(bf16 [N][K]) -------
// Block = 256 threads = 8 waves: 1 M-tile shared, 8 N-tiles (one per wave).
// K-chunk = 128: one TDM per wave per chunk, 4 WMMAs per barrier pair.
__device__ __forceinline__ v8f gemm16x16(const __bf16* A, int lda,
                                         const __bf16* BT, int ldbt,
                                         int m0, int n0, int Kdim,
                                         __bf16* aL, __bf16* bL) {
  const int tid = threadIdx.x, lane = tid & 31, wave = tid >> 5;
  v8f acc = {};
  for (int k0 = 0; k0 < Kdim; k0 += 128) {
    asm volatile("s_wait_dscnt 0" ::: "memory");  // prior LDS reads done
    tdm_load_2d(bL, BT + (size_t)n0 * ldbt + k0,
                (unsigned)ldbt, 16u, (unsigned)ldbt, 128u, 16u);
    if (wave == 0)
      tdm_load_2d(aL, A + (size_t)m0 * lda + k0,
                  (unsigned)lda, 16u, (unsigned)lda, 128u, 16u);
    __builtin_amdgcn_s_wait_tensorcnt(0);
    asm volatile("" ::: "memory");
    __syncthreads();
#pragma unroll
    for (int kc = 0; kc < 4; ++kc) {
      v16bf af = afrag(aL, 128, 32 * kc, lane);
      v16bf bf = bfragS(bL, 128, 0, 32 * kc, lane);
      acc = wmma_bf16(af, bf, acc);
    }
    __syncthreads();
  }
  return acc;
}

// ---- kernel 0a: fp32 -> bf16, coalesced, 4 elems/thread -------------------
__global__ __launch_bounds__(256) void k_cvt4(const float* __restrict__ w,
                                              __bf16* __restrict__ o) {
  int i = blockIdx.x * 256 + threadIdx.x;
  float4 f = ((const float4*)w)[i];
  bf16x4 b = {(__bf16)f.x, (__bf16)f.y, (__bf16)f.z, (__bf16)f.w};
  ((bf16x4*)o)[i] = b;
}

// ---- kernel 0b: fp32 weight [K][N] -> bf16 transposed [N][K] --------------
__global__ __launch_bounds__(256) void k_wconv(const float* __restrict__ w,
                                               __bf16* __restrict__ wt,
                                               int K, int N) {
  int idx = blockIdx.x * 256 + threadIdx.x;   // coalesced read of w
  int k = idx / N, n = idx % N;
  wt[(size_t)n * K + k] = (__bf16)w[idx];
}

// ---- kernel 1: qkv = x @ Wqkv ; split into (q+u), (q+v), k, v^T (bf16) ----
__global__ __launch_bounds__(256) void k_qkv(
    const __bf16* __restrict__ xb, const __bf16* __restrict__ WqkvT,
    const float* __restrict__ u, const float* __restrict__ v,
    __bf16* __restrict__ qu, __bf16* __restrict__ qv,
    __bf16* __restrict__ kk, __bf16* __restrict__ vvT) {
  __shared__ __align__(16) __bf16 aL[16 * 128];
  __shared__ __align__(16) __bf16 bL[8][16 * 128];
  int wave = threadIdx.x >> 5, lane = threadIdx.x & 31;
  int m0 = blockIdx.y * 16;
  int n0 = blockIdx.x * 128 + wave * 16;
  v8f acc = gemm16x16(xb, DIMX, WqkvT, DIMX, m0, n0, DIMX, aL, bL[wave]);

  int n = lane & 15, half = lane >> 4;
  int gcol = n0 + n;
  int sec = gcol >> 10, rem = gcol & 1023;
  int h = rem >> 6, d = rem & 63;
#pragma unroll
  for (int r = 0; r < 8; ++r) {
    int grow = m0 + r + 8 * half;
    int b = grow >> 11, t = grow & (TX - 1);
    float val = acc[r];
    if (sec == 0) {
      size_t idx = ((size_t)(b * HEADSX + h) * TX + t) * HDIMX + d;
      qu[idx] = (__bf16)(val + u[h * HDIMX + d]);
      qv[idx] = (__bf16)(val + v[h * HDIMX + d]);
    } else if (sec == 1) {
      kk[((size_t)(b * HEADSX + h) * TX + t) * HDIMX + d] = (__bf16)val;
    } else {  // value stored transposed [b][h][d][t] (pair-friendly PV frags)
      vvT[((size_t)(b * HEADSX + h) * HDIMX + d) * TX + t] = (__bf16)val;
    }
  }
}

// ---- kernel 2: r_h = r @ Wr -> bf16 [h][t][d] -----------------------------
__global__ __launch_bounds__(256) void k_rproj(
    const __bf16* __restrict__ rb, const __bf16* __restrict__ WrT,
    __bf16* __restrict__ rh) {
  __shared__ __align__(16) __bf16 aL[16 * 128];
  __shared__ __align__(16) __bf16 bL[8][16 * 128];
  int wave = threadIdx.x >> 5, lane = threadIdx.x & 31;
  int m0 = blockIdx.y * 16;
  int n0 = blockIdx.x * 128 + wave * 16;
  v8f acc = gemm16x16(rb, DIMX, WrT, DIMX, m0, n0, DIMX, aL, bL[wave]);

  int n = lane & 15, half = lane >> 4;
  int gcol = n0 + n;
  int h = gcol >> 6, d = gcol & 63;
#pragma unroll
  for (int r8 = 0; r8 < 8; ++r8) {
    int t = m0 + r8 + 8 * half;
    rh[((size_t)h * TX + t) * HDIMX + d] = (__bf16)acc[r8];
  }
}

// ---- kernel 3: fused flash attention with TXL relative shift --------------
// One wave per block; K/V/band tiles double-buffered through the TDM so the
// DMA of key-block kb+1 overlaps the WMMA/softmax of key-block kb.
__global__ __launch_bounds__(32) void k_attn(
    const __bf16* __restrict__ qu, const __bf16* __restrict__ qv,
    const __bf16* __restrict__ kk, const __bf16* __restrict__ vvT,
    const __bf16* __restrict__ rh, const unsigned char* __restrict__ km,
    __bf16* __restrict__ ao) {
  __shared__ __align__(16) __bf16 KT[2][32 * 64];  // K tile  [s][d]
  __shared__ __align__(16) __bf16 VT[2][64 * 32];  // V tile  [d][s]
  __shared__ __align__(16) __bf16 RB[2][48 * 64];  // r_h band [jj][d]
  __shared__ __align__(16) __bf16 PP[16 * 32];     // probs relayout buffer

  const int lane = threadIdx.x;
  const int n = lane & 15, half = lane >> 4;

  int qt = blockIdx.x;                // 0 .. B*H*(T/16)-1
  int tt = qt & 127;                  // T/16 = 128
  int bh = qt >> 7;
  int h = bh & (HEADSX - 1);
  int b = bh >> 4;
  int t0 = tt * 16;

  const __bf16* QUg = qu + ((size_t)(b * HEADSX + h) * TX + t0) * HDIMX;
  const __bf16* QVg = qv + ((size_t)(b * HEADSX + h) * TX + t0) * HDIMX;
  const __bf16* Kg  = kk + (size_t)(b * HEADSX + h) * TX * HDIMX;
  const __bf16* Vg  = vvT + (size_t)(b * HEADSX + h) * HDIMX * TX;
  const __bf16* Rg  = rh + (size_t)h * TX * HDIMX;

  v16bf a_qu0 = afrag(QUg, HDIMX, 0, lane), a_qu1 = afrag(QUg, HDIMX, 32, lane);
  v16bf a_qv0 = afrag(QVg, HDIMX, 0, lane), a_qv1 = afrag(QVg, HDIMX, 32, lane);

  float m_run[8], l_run[8];
  v8f o0 = {}, o1 = {}, o2 = {}, o3 = {};
#pragma unroll
  for (int r8 = 0; r8 < 8; ++r8) { m_run[r8] = -1e30f; l_run[r8] = 0.f; }

  const int nkb = (t0 + 15) / 32 + 1;

  // issue the 3 tensor loads for key block kb into buffer buf
  auto issue = [&](int kb, int buf) {
    const int s0 = kb * 32;
    const int jbase = s0 - t0 + TX - 16;  // always >= 0; OOB rows -> zero
    tdm_load_2d(KT[buf], Kg + (size_t)s0 * HDIMX, 64u, 32u, 64u, 64u, 32u);
    tdm_load_2d(VT[buf], Vg + s0, (unsigned)TX, 64u, (unsigned)TX, 32u, 64u);
    tdm_load_2d(RB[buf], Rg + (size_t)jbase * HDIMX,
                64u, (unsigned)(TX - jbase), 64u, 64u, 48u);
  };
  issue(0, 0);

  for (int kb = 0; kb < nkb; ++kb) {
    const int s0 = kb * 32;
    const int cur = kb & 1;
    if (kb + 1 < nkb) {
      asm volatile("s_wait_dscnt 0" ::: "memory");  // old reads of buf cur^1 done
      issue(kb + 1, cur ^ 1);
      __builtin_amdgcn_s_wait_tensorcnt(3);         // block kb's 3 tiles ready
    } else {
      __builtin_amdgcn_s_wait_tensorcnt(0);
    }
    asm volatile("" ::: "memory");
    const __bf16* KTc = KT[cur];
    const __bf16* VTc = VT[cur];
    const __bf16* RBc = RB[cur];

    // ---- AC = (q+u) @ K^T : two 16x16 score tiles, K-dim = 64 ----
    v8f sc0 = {}, sc1 = {};
    sc0 = wmma_bf16(a_qu0, bfragS(KTc, 64, 0, 0, lane), sc0);
    sc0 = wmma_bf16(a_qu1, bfragS(KTc, 64, 0, 32, lane), sc0);
    sc1 = wmma_bf16(a_qu0, bfragS(KTc, 64, 16, 0, lane), sc1);
    sc1 = wmma_bf16(a_qu1, bfragS(KTc, 64, 16, 32, lane), sc1);

    // ---- BD band: P[t,jj] = (q+v) . r_h[jbase+jj], jj in [0,48) ----
    v8f pb0 = {}, pb1 = {}, pb2 = {};
    pb0 = wmma_bf16(a_qv0, bfragS(RBc, 64, 0, 0, lane), pb0);
    pb0 = wmma_bf16(a_qv1, bfragS(RBc, 64, 0, 32, lane), pb0);
    pb1 = wmma_bf16(a_qv0, bfragS(RBc, 64, 16, 0, lane), pb1);
    pb1 = wmma_bf16(a_qv1, bfragS(RBc, 64, 16, 32, lane), pb1);
    pb2 = wmma_bf16(a_qv0, bfragS(RBc, 64, 32, 0, lane), pb2);
    pb2 = wmma_bf16(a_qv1, bfragS(RBc, 64, 32, 32, lane), pb2);

    // ---- rel-shift diagonal extraction + causal/key mask + scale ----
    unsigned char km0 = km[(size_t)b * TX + s0 + n];
    unsigned char km1 = km[(size_t)b * TX + s0 + 16 + n];
#pragma unroll
    for (int r8 = 0; r8 < 8; ++r8) {
      int M = r8 + 8 * half;
      {  // tile 0 (keys s0..s0+15)
        int jj = n + 15 - M;
        int src = (jj & 15) | (lane & 16);
        float b0 = shflf(pb0[r8], src), b1 = shflf(pb1[r8], src), b2 = shflf(pb2[r8], src);
        float bd = (jj < 16) ? b0 : (jj < 32) ? b1 : b2;
        float s = (sc0[r8] + bd) * 0.125f;
        bool dead = ((s0 + n) > (t0 + M)) || km0;
        sc0[r8] = dead ? -1e30f : s;
      }
      {  // tile 1 (keys s0+16..s0+31)
        int jj = n + 31 - M;
        int src = (jj & 15) | (lane & 16);
        float b0 = shflf(pb0[r8], src), b1 = shflf(pb1[r8], src), b2 = shflf(pb2[r8], src);
        float bd = (jj < 16) ? b0 : (jj < 32) ? b1 : b2;
        float s = (sc1[r8] + bd) * 0.125f;
        bool dead = ((s0 + 16 + n) > (t0 + M)) || km1;
        sc1[r8] = dead ? -1e30f : s;
      }
    }

    // ---- online softmax (16-lane-half reductions via ds_swizzle) ----
#pragma unroll
    for (int r8 = 0; r8 < 8; ++r8) {
      float mv = fmaxf(sc0[r8], sc1[r8]);
      mv = fmaxf(mv, SWZ_XOR(mv, 1));
      mv = fmaxf(mv, SWZ_XOR(mv, 2));
      mv = fmaxf(mv, SWZ_XOR(mv, 4));
      mv = fmaxf(mv, SWZ_XOR(mv, 8));
      float mn = fmaxf(m_run[r8], mv);
      float al = __expf(m_run[r8] - mn);
      m_run[r8] = mn;
      float p0 = (sc0[r8] < -1e29f) ? 0.f : __expf(sc0[r8] - mn);
      float p1 = (sc1[r8] < -1e29f) ? 0.f : __expf(sc1[r8] - mn);
      sc0[r8] = p0; sc1[r8] = p1;
      float rs = p0 + p1;
      rs += SWZ_XOR(rs, 1);
      rs += SWZ_XOR(rs, 2);
      rs += SWZ_XOR(rs, 4);
      rs += SWZ_XOR(rs, 8);
      l_run[r8] = l_run[r8] * al + rs;
      o0[r8] *= al; o1[r8] *= al; o2[r8] *= al; o3[r8] *= al;
    }

    // ---- P (C-layout) -> bf16 A-fragment via LDS relayout ----
#pragma unroll
    for (int r8 = 0; r8 < 8; ++r8) {
      int M = r8 + 8 * half;
      PP[M * 32 + n]      = (__bf16)sc0[r8];
      PP[M * 32 + 16 + n] = (__bf16)sc1[r8];
    }
    asm volatile("s_wait_dscnt 0" ::: "memory");
    v16bf ap = afrag(PP, 32, 0, lane);

    // ---- O += P @ V : 4 d-tiles (VT is [d][s] -> pair-contiguous) ----
    o0 = wmma_bf16(ap, bfragS(VTc, 32, 0, 0, lane), o0);
    o1 = wmma_bf16(ap, bfragS(VTc, 32, 16, 0, lane), o1);
    o2 = wmma_bf16(ap, bfragS(VTc, 32, 32, 0, lane), o2);
    o3 = wmma_bf16(ap, bfragS(VTc, 32, 48, 0, lane), o3);
  }

  // ---- normalize (nan_to_num: fully-masked row -> 0) and store bf16 ----
#pragma unroll
  for (int r8 = 0; r8 < 8; ++r8) {
    float inv = (l_run[r8] > 0.f) ? (1.f / l_run[r8]) : 0.f;
    int t = t0 + r8 + 8 * half;
    __bf16* dst = ao + ((size_t)b * TX + t) * DIMX + h * HDIMX;
    dst[n]      = (__bf16)(o0[r8] * inv);
    dst[16 + n] = (__bf16)(o1[r8] * inv);
    dst[32 + n] = (__bf16)(o2[r8] * inv);
    dst[48 + n] = (__bf16)(o3[r8] * inv);
  }
}

// ---- kernel 4: out = AO(bf16) @ Wout + bout -> fp32 -----------------------
__global__ __launch_bounds__(256) void k_outproj(
    const __bf16* __restrict__ ao, const __bf16* __restrict__ WoutT,
    const float* __restrict__ bout, float* __restrict__ out) {
  __shared__ __align__(16) __bf16 aL[16 * 128];
  __shared__ __align__(16) __bf16 bL[8][16 * 128];
  int wave = threadIdx.x >> 5, lane = threadIdx.x & 31;
  int m0 = blockIdx.y * 16;
  int n0 = blockIdx.x * 128 + wave * 16;
  v8f acc = gemm16x16(ao, DIMX, WoutT, DIMX, m0, n0, DIMX, aL, bL[wave]);

  int n = lane & 15, half = lane >> 4;
  int gcol = n0 + n;
#pragma unroll
  for (int r8 = 0; r8 < 8; ++r8) {
    int grow = m0 + r8 + 8 * half;
    out[(size_t)grow * DIMX + gcol] = acc[r8] + bout[gcol];
  }
}

// ---------------------------------------------------------------------------
extern "C" void kernel_launch(void* const* d_in, const int* in_sizes, int n_in,
                              void* d_out, int out_size, void* d_ws, size_t ws_size,
                              hipStream_t stream) {
  (void)in_sizes; (void)n_in; (void)out_size; (void)ws_size;
  const float* x    = (const float*)d_in[0];
  const float* r    = (const float*)d_in[1];
  const unsigned char* km = (const unsigned char*)d_in[2];
  const float* Wqkv = (const float*)d_in[3];
  const float* Wr   = (const float*)d_in[4];
  const float* u    = (const float*)d_in[5];
  const float* v    = (const float*)d_in[6];
  const float* Wout = (const float*)d_in[7];
  const float* bout = (const float*)d_in[8];
  float* out = (float*)d_out;

  char* ws = (char*)d_ws;
  const size_t per  = (size_t)BX * HEADSX * TX * HDIMX * 2;  // 8 MB (bf16)
  const size_t rhsz = (size_t)HEADSX * TX * HDIMX * 2;       // 4 MB
  const size_t aosz = (size_t)BX * TX * DIMX * 2;            // 8 MB
  __bf16* qu    = (__bf16*)(ws);
  __bf16* qv    = (__bf16*)(ws + per);
  __bf16* kk    = (__bf16*)(ws + 2 * per);
  __bf16* vvT   = (__bf16*)(ws + 3 * per);
  __bf16* rh    = (__bf16*)(ws + 4 * per);
  __bf16* ao    = (__bf16*)(ws + 4 * per + rhsz);
  char*   wsw   = ws + 4 * per + rhsz + aosz;
  __bf16* WqkvT = (__bf16*)(wsw);                                  // 6 MB
  __bf16* WrT   = (__bf16*)(wsw + (size_t)DIMX * 3 * DIMX * 2);    // 2 MB
  __bf16* WoutT = (__bf16*)(wsw + (size_t)DIMX * 4 * DIMX * 2);    // 2 MB
  __bf16* xb    = (__bf16*)(wsw + (size_t)DIMX * 5 * DIMX * 2);    // 8 MB
  __bf16* rb    = (__bf16*)((char*)xb + (size_t)BX * TX * DIMX * 2);  // 4 MB

  // 0) one-shot conversions: activations bf16, weights bf16-transposed [N][K]
  k_cvt4<<<(BX * TX * DIMX / 4) / 256, 256, 0, stream>>>(x, xb);
  k_cvt4<<<(TX * DIMX / 4) / 256, 256, 0, stream>>>(r, rb);
  k_wconv<<<(DIMX * 3 * DIMX) / 256, 256, 0, stream>>>(Wqkv, WqkvT, DIMX, 3 * DIMX);
  k_wconv<<<(DIMX * DIMX) / 256, 256, 0, stream>>>(Wr, WrT, DIMX, DIMX);
  k_wconv<<<(DIMX * DIMX) / 256, 256, 0, stream>>>(Wout, WoutT, DIMX, DIMX);

  // 1) qkv projection + bias split (M=4096, N=3072, K=1024)
  k_qkv<<<dim3((3 * DIMX) / 128, (BX * TX) / 16), 256, 0, stream>>>(
      xb, WqkvT, u, v, qu, qv, kk, vvT);
  // 2) r projection (M=2048, N=1024, K=1024)
  k_rproj<<<dim3(DIMX / 128, TX / 16), 256, 0, stream>>>(rb, WrT, rh);
  // 3) fused flash attention: one wave (16 query rows) per block
  k_attn<<<dim3(BX * HEADSX * (TX / 16)), 32, 0, stream>>>(
      qu, qv, kk, vvT, rh, km, ao);
  // 4) output projection (M=4096, N=1024, K=1024)
  k_outproj<<<dim3(DIMX / 128, (BX * TX) / 16), 256, 0, stream>>>(
      ao, WoutT, bout, out);
}
